// CompactBilinearPooling_61083024884165
// MI455X (gfx1250) — matrix operand order repaired and checked by reading, compiled, not run on hardware
//
#include <hip/hip_runtime.h>
#include <hip/hip_bf16.h>

// Compact bilinear pooling, reformulated:
//   out[b,d] = sum_{c1,c2: (h1[c1]+h2[c2]) mod D == d} s1[c1]*s2[c2]*G[b,c1,c2]
//   G[b] = X1_b (512x196) @ X2_b^T (196x512)   -- batched GEMM via bf16 WMMA
// Fused kernel: GEMM tiles -> scatter into LDS accum[8192] -> global atomic add.

#define OUT_DIM 8192
#define B_SZ    32
#define C_SZ    512
#define HW      196          // 14*14
#define KCHUNKS 7            // 7*32 = 224 >= 196 (tail masked)

typedef __attribute__((ext_vector_type(16))) __bf16 v16bf;
typedef __attribute__((ext_vector_type(8)))  float  v8f;

__global__ void cbp_zero_out(float* __restrict__ out, int n) {
    int i = blockIdx.x * blockDim.x + threadIdx.x;
    if (i < n) out[i] = 0.0f;
}

__launch_bounds__(256)
__global__ void cbp_fused(const float* __restrict__ x1,
                          const float* __restrict__ x2,
                          const int* __restrict__ rand_h1,
                          const int* __restrict__ rand_s1,
                          const int* __restrict__ rand_h2,
                          const int* __restrict__ rand_s2,
                          float* __restrict__ out)
{
    __shared__ float accum[OUT_DIM];   // 32 KB per-block sketch accumulator
    __shared__ int   h1s[C_SZ];
    __shared__ int   h2s[C_SZ];
    __shared__ float s1s[C_SZ];        // signs as +/-1.0f, folded into operands
    __shared__ float s2s[C_SZ];

    const int tid     = threadIdx.x;
    const int b       = blockIdx.x >> 3;      // batch index
    const int split   = blockIdx.x & 7;       // 8 splits of 64 c1-rows
    const int c1_base = split * 64;

    for (int i = tid; i < OUT_DIM; i += 256) accum[i] = 0.0f;
    for (int i = tid; i < C_SZ; i += 256) {
        h1s[i] = rand_h1[i];
        h2s[i] = rand_h2[i];
        s1s[i] = (float)(2 * rand_s1[i] - 1);
        s2s[i] = (float)(2 * rand_s2[i] - 1);
    }
    __syncthreads();

    const int wave  = tid >> 5;   // 0..7
    const int lane  = tid & 31;
    const int lhalf = lane >> 4;  // which K-half this lane owns in A/B fragments
    const int l16   = lane & 15;

    const float* base1 = x1 + (size_t)b * C_SZ * HW;
    const float* base2 = x2 + (size_t)b * C_SZ * HW;

    // 4 i-tiles (64 c1 rows) x 32 j-tiles (512 c2 cols) = 128 tiles; 16 per wave.
    for (int t = wave; t < 128; t += 8) {
        const int ti = t >> 5;                      // 0..3
        const int tj = t & 31;                      // 0..31
        const int c1 = c1_base + ti * 16 + l16;     // A row owned by this lane
        const int c2 = tj * 16 + l16;               // B col owned by this lane
        const float sa = s1s[c1];
        const float sb = s2s[c2];
        const float* rowA = base1 + (size_t)c1 * HW;
        const float* rowB = base2 + (size_t)c2 * HW;
        const int abase = lhalf * 8;    // A: lanes 0-15 K=0..7,16..23; 16-31: +8
        const int bbase = lhalf * 8;    // B: lanes 0-15 K=0..7,16..23; 16-31: +8

        v8f acc = {};
        for (int kc = 0; kc < KCHUNKS; ++kc) {
            const int k0 = kc * 32;
            v16bf av, bv;
            if (kc < 6) {
                // Fully in-range chunk: aligned float4 loads (32B-aligned offsets).
                const float4 a0 = *(const float4*)(rowA + k0 + abase);
                const float4 a1 = *(const float4*)(rowA + k0 + abase + 4);
                const float4 a2 = *(const float4*)(rowA + k0 + abase + 16);
                const float4 a3 = *(const float4*)(rowA + k0 + abase + 20);
                av[0]  = (__bf16)(a0.x * sa); av[1]  = (__bf16)(a0.y * sa);
                av[2]  = (__bf16)(a0.z * sa); av[3]  = (__bf16)(a0.w * sa);
                av[4]  = (__bf16)(a1.x * sa); av[5]  = (__bf16)(a1.y * sa);
                av[6]  = (__bf16)(a1.z * sa); av[7]  = (__bf16)(a1.w * sa);
                av[8]  = (__bf16)(a2.x * sa); av[9]  = (__bf16)(a2.y * sa);
                av[10] = (__bf16)(a2.z * sa); av[11] = (__bf16)(a2.w * sa);
                av[12] = (__bf16)(a3.x * sa); av[13] = (__bf16)(a3.y * sa);
                av[14] = (__bf16)(a3.z * sa); av[15] = (__bf16)(a3.w * sa);
                const float4 b0 = *(const float4*)(rowB + k0 + bbase);
                const float4 b1 = *(const float4*)(rowB + k0 + bbase + 4);
                const float4 b2 = *(const float4*)(rowB + k0 + bbase + 16);
                const float4 b3 = *(const float4*)(rowB + k0 + bbase + 20);
                bv[0]  = (__bf16)(b0.x * sb); bv[1]  = (__bf16)(b0.y * sb);
                bv[2]  = (__bf16)(b0.z * sb); bv[3]  = (__bf16)(b0.w * sb);
                bv[4]  = (__bf16)(b1.x * sb); bv[5]  = (__bf16)(b1.y * sb);
                bv[6]  = (__bf16)(b1.z * sb); bv[7]  = (__bf16)(b1.w * sb);
                bv[8]  = (__bf16)(b2.x * sb); bv[9]  = (__bf16)(b2.y * sb);
                bv[10] = (__bf16)(b2.z * sb); bv[11] = (__bf16)(b2.w * sb);
                bv[12] = (__bf16)(b3.x * sb); bv[13] = (__bf16)(b3.y * sb);
                bv[14] = (__bf16)(b3.z * sb); bv[15] = (__bf16)(b3.w * sb);
            } else {
                // Tail chunk (K = 192..223, only 192..195 valid): masked loads.
                #pragma unroll
                for (int e = 0; e < 16; ++e) {
                    const int ka = k0 + abase + ((e < 8) ? e : (8 + e)); // +16 jump
                    av[e] = (__bf16)((ka < HW) ? rowA[ka] * sa : 0.0f);
                    const int kb = k0 + bbase + ((e < 8) ? e : (8 + e));
                    bv[e] = (__bf16)((kb < HW) ? rowB[kb] * sb : 0.0f);
                }
            }
            // D = A(16x32 bf16) x B(32x16 bf16) + C(16x16 f32)
            acc = __builtin_amdgcn_wmma_f32_16x16x32_bf16(
                      false, av, false, bv, (short)0, acc, false, false);
        }

        // Scatter the 16x16 tile into the LDS sketch accumulator.
        // C/D layout: lane l, vgpr v -> M = v + 8*(l>=16), N = l&15.
        const int c2g = tj * 16 + l16;
        const int h2v = h2s[c2g];
        #pragma unroll
        for (int v = 0; v < 8; ++v) {
            const int c1g = c1_base + ti * 16 + v + 8 * lhalf;
            int d = h1s[c1g] + h2v;                    // both < 8192
            d = (d >= OUT_DIM) ? (d - OUT_DIM) : d;    // circular wrap
            atomicAdd(&accum[d], acc[v]);              // ds_add_f32
        }
    }

    __syncthreads();
    float* orow = out + (size_t)b * OUT_DIM;
    for (int i = tid; i < OUT_DIM; i += 256)
        atomicAdd(&orow[i], accum[i]);                 // combine 8 c1-splits
}

extern "C" void kernel_launch(void* const* d_in, const int* in_sizes, int n_in,
                              void* d_out, int out_size, void* d_ws, size_t ws_size,
                              hipStream_t stream) {
    const float* bottom1 = (const float*)d_in[0];
    const float* bottom2 = (const float*)d_in[1];
    const int*   rand_h1 = (const int*)d_in[2];
    const int*   rand_s1 = (const int*)d_in[3];
    const int*   rand_h2 = (const int*)d_in[4];
    const int*   rand_s2 = (const int*)d_in[5];
    float* out = (float*)d_out;

    const int n = B_SZ * OUT_DIM;
    cbp_zero_out<<<(n + 255) / 256, 256, 0, stream>>>(out, n);
    cbp_fused<<<B_SZ * 8, 256, 0, stream>>>(bottom1, bottom2,
                                            rand_h1, rand_s1,
                                            rand_h2, rand_s2, out);
}
